// TransformerLanguageModel_2611340116120
// MI455X (gfx1250) — compile-verified, matrix operand order checked by reading, not run on hardware
//
#include <hip/hip_runtime.h>
#include <hip/hip_bf16.h>

// ---------------------------------------------------------------------------
// GPT forward on gfx1250 (MI455X): bf16 WMMA GEMMs (double-buffered async-LDS
// staging) + register flash-attention with prefetched fragments.
// ---------------------------------------------------------------------------

typedef __attribute__((ext_vector_type(16))) __bf16 v16bf;
typedef __attribute__((ext_vector_type(8)))  __bf16 v8bf;
typedef __attribute__((ext_vector_type(8)))  float  v8f;

#define GPT_B  4
#define GPT_T  1024
#define GPT_D  1024
#define GPT_H  16
#define GPT_HS 64
#define GPT_L  4
#define GPT_V  32000
#define GPT_F  4096
#define ROWS   (GPT_B * GPT_T)   // 4096 token rows

__device__ __forceinline__ v8f wmma_bf(v16bf a, v16bf b, v8f c) {
  // D = A(16x32 bf16) * B(32x16 bf16) + C(16x16 f32)
  return __builtin_amdgcn_wmma_f32_16x16x32_bf16(false, a, false, b,
                                                 (short)0, c, false, false);
}

__device__ __forceinline__ v16bf combine8(v8bf lo, v8bf hi) {
  v16bf r;
#pragma unroll
  for (int i = 0; i < 8; ++i) { r[i] = lo[i]; r[i + 8] = hi[i]; }
  return r;
}

__device__ __forceinline__ v8f zero8() {
  v8f z;
#pragma unroll
  for (int i = 0; i < 8; ++i) z[i] = 0.f;
  return z;
}

// CDNA5 async global->LDS copy (16B per lane), tracked by ASYNCcnt.
__device__ __forceinline__ void async_copy_b128(unsigned lds_byte_addr,
                                                unsigned long long gaddr) {
  asm volatile("global_load_async_to_lds_b128 %0, %1, off"
               :: "v"(lds_byte_addr), "v"(gaddr) : "memory");
}
__device__ __forceinline__ void wait_async0() {
  asm volatile("s_wait_asynccnt 0" ::: "memory");
}

// ---------------------------------------------------------------------------
// Weight prep: fp32 -> bf16 casts
// ---------------------------------------------------------------------------
__global__ void cvt_f32_bf16(const float* __restrict__ in,
                             __bf16* __restrict__ out, long long n) {
  long long i = (long long)blockIdx.x * blockDim.x + threadIdx.x;
  if (i < n) out[i] = (__bf16)in[i];
}

// Pack Wq/Wk/Wv [L,H,D,HS] -> per-layer row-major [D, 3*D] bf16 (heads fused).
__global__ void pack_qkv_w(const float* __restrict__ Wq,
                           const float* __restrict__ Wk,
                           const float* __restrict__ Wv,
                           __bf16* __restrict__ out) {
  long long i = (long long)blockIdx.x * blockDim.x + threadIdx.x;
  const long long total = (long long)GPT_L * GPT_D * (3 * GPT_D);
  if (i >= total) return;
  int n = (int)(i % (3 * GPT_D));
  int k = (int)((i / (3 * GPT_D)) % GPT_D);
  int l = (int)(i / ((long long)(3 * GPT_D) * GPT_D));
  int which = n / GPT_D;           // 0=q 1=k 2=v
  int nn = n % GPT_D;
  int h = nn / GPT_HS, hs = nn % GPT_HS;
  const float* W = (which == 0) ? Wq : (which == 1) ? Wk : Wv;
  out[i] = (__bf16)W[(((long long)l * GPT_H + h) * GPT_D + k) * GPT_HS + hs];
}

// ---------------------------------------------------------------------------
// Embedding: x[r,d] = tok_emb[idx[r],d] + pos_emb[r%T,d]
// ---------------------------------------------------------------------------
__global__ void embed_kernel(const int* __restrict__ idx,
                             const float* __restrict__ tok,
                             const float* __restrict__ pos,
                             float* __restrict__ x) {
  long long i = (long long)blockIdx.x * blockDim.x + threadIdx.x;
  if (i >= (long long)ROWS * GPT_D) return;
  int d = (int)(i & (GPT_D - 1));
  int r = (int)(i >> 10);
  int t = r & (GPT_T - 1);
  int tk = idx[r];
  x[i] = tok[(long long)tk * GPT_D + d] + pos[(long long)t * GPT_D + d];
}

// ---------------------------------------------------------------------------
// LayerNorm over D=1024, fp32 in -> bf16 out
// ---------------------------------------------------------------------------
__global__ __launch_bounds__(128)
void ln_bf16(const float* __restrict__ x, const float* __restrict__ s,
             const float* __restrict__ b, __bf16* __restrict__ out) {
  int row = blockIdx.x, tid = threadIdx.x;
  const float* xr = x + (long long)row * GPT_D;
  __shared__ float red[128];
  float acc = 0.f;
  for (int d = tid; d < GPT_D; d += 128) acc += xr[d];
  red[tid] = acc; __syncthreads();
  for (int st = 64; st > 0; st >>= 1) {
    if (tid < st) red[tid] += red[tid + st];
    __syncthreads();
  }
  float mean = red[0] * (1.f / GPT_D);
  __syncthreads();
  acc = 0.f;
  for (int d = tid; d < GPT_D; d += 128) { float t = xr[d] - mean; acc += t * t; }
  red[tid] = acc; __syncthreads();
  for (int st = 64; st > 0; st >>= 1) {
    if (tid < st) red[tid] += red[tid + st];
    __syncthreads();
  }
  float rstd = rsqrtf(red[0] * (1.f / GPT_D) + 1e-5f);
  for (int d = tid; d < GPT_D; d += 128)
    out[(long long)row * GPT_D + d] = (__bf16)((xr[d] - mean) * rstd * s[d] + b[d]);
}

// ---------------------------------------------------------------------------
// Tiled bf16 GEMM, compile-time specialized epilogue, double-buffered LDS.
//   C[M,N] = act(A[M,K] @ B[K,N] [+ bias] [+ resid])
// Block 128x128, 8 waves (4x2), per-wave 32x64 = 8 WMMA accumulators.
// A tile staged via CDNA5 global_load_async_to_lds_b128 (ASYNCcnt) into the
// ping-pong buffer for step kt+1 while step kt's WMMAs run; one barrier/step.
// B tile staged transposed so fragments load as two ds_load_b128 per lane.
// ---------------------------------------------------------------------------
template <int BIAS, int RELU, int RESID, int OUTF>
__global__ __launch_bounds__(256)
void gemm_bf16(const __bf16* __restrict__ A, const __bf16* __restrict__ Bm,
               int M, int N, int K,
               const float* __restrict__ bias, const float* __restrict__ resid,
               float* __restrict__ outF, __bf16* __restrict__ outB) {
  __shared__ __align__(16) __bf16 As[2][128 * 32];
  __shared__ __align__(16) __bf16 Bs[2][128 * 32];   // transposed: Bs[n][k]
  const int tid = threadIdx.x;
  const int lane = tid & 31, w = tid >> 5;
  const int wy = w & 3, wx = w >> 2;
  const int m0 = blockIdx.y * 128, n0 = blockIdx.x * 128;
  const int hh = lane >> 4, ln15 = lane & 15;

  v8f acc[2][4];
#pragma unroll
  for (int i = 0; i < 2; ++i)
#pragma unroll
    for (int j = 0; j < 4; ++j) acc[i][j] = zero8();

  // A staging geometry: 512 16B chunks, 2 consecutive per thread (same row).
  const int aq = tid * 2;
  const int arow = aq >> 2, acol = (aq & 3) * 8;   // element col offset
  const int bnn = tid >> 1, bkh = (tid & 1) * 16;

  auto stage = [&](int buf, int k0) {
    unsigned long long ga =
        (unsigned long long)(A + (long long)(m0 + arow) * K + k0 + acol);
    unsigned la = (unsigned)(unsigned long long)(&As[buf][arow * 32 + acol]);
    async_copy_b128(la, ga);
    async_copy_b128(la + 16, ga + 16);
    const __bf16* src = Bm + (long long)(k0 + bkh) * N + n0 + bnn;
    __bf16* dst = &Bs[buf][bnn * 32 + bkh];
#pragma unroll
    for (int kk = 0; kk < 16; ++kk)
      dst[kk] = src[(long long)kk * N];
  };

  const int NK = K >> 5;
  stage(0, 0);
  for (int kt = 0; kt < NK; ++kt) {
    wait_async0();     // previous stage's async LDS writes (this wave) landed
    __syncthreads();   // all staging visible; all reads of next buffer done
    const int cur = kt & 1;
    if (kt + 1 < NK) stage(cur ^ 1, (kt + 1) * 32);

    const __bf16* Ac = As[cur];
    const __bf16* Bc = Bs[cur];
    v16bf af[2];
#pragma unroll
    for (int i = 0; i < 2; ++i) {
      const __bf16* p = Ac + (wy * 32 + i * 16 + ln15) * 32;
      af[i] = combine8(*(const v8bf*)(p + 8 * hh), *(const v8bf*)(p + 16 + 8 * hh));
    }
    v16bf bfr[4];
#pragma unroll
    for (int j = 0; j < 4; ++j) {
      const __bf16* p = Bc + (wx * 64 + j * 16 + ln15) * 32;
      bfr[j] = combine8(*(const v8bf*)(p + 8 * hh), *(const v8bf*)(p + 16 + 8 * hh));
    }
#pragma unroll
    for (int i = 0; i < 2; ++i)
#pragma unroll
      for (int j = 0; j < 4; ++j)
        acc[i][j] = wmma_bf(af[i], bfr[j], acc[i][j]);
  }

  // Branch-free specialized epilogue.
  float biasv[4];
  if (BIAS) {
#pragma unroll
    for (int j = 0; j < 4; ++j)
      biasv[j] = bias[n0 + wx * 64 + j * 16 + ln15];
  }
#pragma unroll
  for (int i = 0; i < 2; ++i)
#pragma unroll
    for (int j = 0; j < 4; ++j) {
      const int gn = n0 + wx * 64 + j * 16 + ln15;
      const int gmb = m0 + wy * 32 + i * 16 + 8 * hh;
#pragma unroll
      for (int r = 0; r < 8; ++r) {
        const long long o = (long long)(gmb + r) * N + gn;
        float v = acc[i][j][r];
        if (BIAS)  v += biasv[j];
        if (RELU)  v = fmaxf(v, 0.f);
        if (RESID) v += resid[o];
        if (OUTF)  outF[o] = v;
        else       outB[o] = (__bf16)v;
      }
    }
}

// ---------------------------------------------------------------------------
// Flash attention, one wave per (b, h, 16-query tile). All-WMMA:
//   S^T(16k x 16q) = K(16x32) * Q^T(32x16)       [2 wmma per key sub-tile]
//   O^T(hs x 16q) += V^T(16x32) * P^T(32x16)     [4 wmma per 32-key block]
// All K/V fragments for a 32-key block are prefetched before any compute so
// the scalar V gathers overlap the S^T WMMAs and softmax VALU work.
// C-layout of S^T == B-layout of P^T => zero-shuffle handoff; softmax stats
// need one __shfl_xor(16) per reduction.
// ---------------------------------------------------------------------------
__global__ __launch_bounds__(32)
void attn_kernel(const __bf16* __restrict__ qkv, __bf16* __restrict__ attnout) {
  const int lane = threadIdx.x;
  const int bh = blockIdx.x;
  const int b = bh / GPT_H, h = bh % GPT_H;
  const int q0 = blockIdx.y * 16;
  const int nq = lane & 15, hh = lane >> 4;
  const int LDQ = 3 * GPT_D;
  const float scale = 0.125f;  // HS^-0.5

  // Q^T fragments (B-operand), pre-scaled
  v16bf qb[2];
  {
    const __bf16* qrow = qkv + (long long)(b * GPT_T + q0 + nq) * LDQ + h * GPT_HS;
#pragma unroll
    for (int t = 0; t < 2; ++t) {
      v8bf lo = *(const v8bf*)(qrow + 32 * t + 8 * hh);
      v8bf hi = *(const v8bf*)(qrow + 32 * t + 16 + 8 * hh);
#pragma unroll
      for (int e = 0; e < 8; ++e) {
        qb[t][e]     = (__bf16)((float)lo[e] * scale);
        qb[t][e + 8] = (__bf16)((float)hi[e] * scale);
      }
    }
  }

  v8f o[4];
#pragma unroll
  for (int t = 0; t < 4; ++t) o[t] = zero8();
  float m_i = -3.0e38f, l_i = 0.f;
  const int kmax = q0 + 15;

  for (int kb = 0; kb * 32 <= kmax; ++kb) {
    const int kbase = kb * 32;

    // ---- prefetch ALL operands for this 32-key block ----
    v16bf ka[2][2];
#pragma unroll
    for (int t2 = 0; t2 < 2; ++t2) {
      const __bf16* krow = qkv +
          (long long)(b * GPT_T + kbase + t2 * 16 + nq) * LDQ + GPT_D + h * GPT_HS;
#pragma unroll
      for (int f = 0; f < 2; ++f)
        ka[t2][f] = combine8(*(const v8bf*)(krow + 32 * f + 8 * hh),
                             *(const v8bf*)(krow + 32 * f + 16 + 8 * hh));
    }
    v16bf va[4];
#pragma unroll
    for (int tt = 0; tt < 4; ++tt) {
#pragma unroll
      for (int e = 0; e < 16; ++e) {
        int kk = (e < 8) ? (8 * hh + e) : (16 + 8 * hh + (e - 8));
        va[tt][e] = qkv[(long long)(b * GPT_T + kbase + kk) * LDQ + 2 * GPT_D +
                        h * GPT_HS + tt * 16 + nq];
      }
    }

    // ---- S^T = K * Q^T, causal mask ----
    v8f st[2];
#pragma unroll
    for (int t2 = 0; t2 < 2; ++t2) {
      const int k0 = kbase + t2 * 16;
      v8f c = zero8();
      c = wmma_bf(ka[t2][0], qb[0], c);
      c = wmma_bf(ka[t2][1], qb[1], c);
#pragma unroll
      for (int r = 0; r < 8; ++r)
        if (k0 + r + 8 * hh > q0 + nq) c[r] = -3.0e38f;
      st[t2] = c;
    }

    // ---- online softmax (lane pairs L/L+16 combined via xor-16) ----
    float mx = -3.0e38f;
#pragma unroll
    for (int t2 = 0; t2 < 2; ++t2)
#pragma unroll
      for (int r = 0; r < 8; ++r) mx = fmaxf(mx, st[t2][r]);
    mx = fmaxf(mx, __shfl_xor(mx, 16, 32));
    float newm = fmaxf(m_i, mx);
    float alpha = __expf(m_i - newm);
    float rs = 0.f;
#pragma unroll
    for (int t2 = 0; t2 < 2; ++t2)
#pragma unroll
      for (int r = 0; r < 8; ++r) {
        float p = __expf(st[t2][r] - newm);
        st[t2][r] = p;
        rs += p;
      }
    rs += __shfl_xor(rs, 16, 32);
    l_i = l_i * alpha + rs;
    m_i = newm;
#pragma unroll
    for (int t = 0; t < 4; ++t)
#pragma unroll
      for (int r = 0; r < 8; ++r) o[t][r] *= alpha;

    // pack P^T: C-layout of st[0]/st[1] maps 1:1 onto B-operand halves
    v16bf pb;
#pragma unroll
    for (int e = 0; e < 8; ++e) {
      pb[e]     = (__bf16)st[0][e];
      pb[e + 8] = (__bf16)st[1][e];
    }

    // ---- O^T += V^T * P^T (4 hs-tiles of 16) ----
#pragma unroll
    for (int tt = 0; tt < 4; ++tt)
      o[tt] = wmma_bf(va[tt], pb, o[tt]);
  }

  const float inv = 1.f / l_i;
#pragma unroll
  for (int tt = 0; tt < 4; ++tt)
#pragma unroll
    for (int r = 0; r < 8; ++r) {
      int hsd = tt * 16 + r + 8 * hh;   // O^T row = head-dim
      attnout[(long long)(b * GPT_T + q0 + nq) * GPT_D + h * GPT_HS + hsd] =
          (__bf16)(o[tt][r] * inv);
    }
}

// ---------------------------------------------------------------------------
// Host orchestration
// ---------------------------------------------------------------------------
extern "C" void kernel_launch(void* const* d_in, const int* in_sizes, int n_in,
                              void* d_out, int out_size, void* d_ws, size_t ws_size,
                              hipStream_t stream) {
  const int*   idx   = (const int*)  d_in[0];
  const float* tok   = (const float*)d_in[1];
  const float* pos   = (const float*)d_in[2];
  const float* Wq    = (const float*)d_in[3];
  const float* Wk    = (const float*)d_in[4];
  const float* Wv    = (const float*)d_in[5];
  const float* Wo    = (const float*)d_in[6];
  const float* bo    = (const float*)d_in[7];
  const float* W1    = (const float*)d_in[8];
  const float* b1    = (const float*)d_in[9];
  const float* W2    = (const float*)d_in[10];
  const float* b2    = (const float*)d_in[11];
  const float* ln1_s = (const float*)d_in[12];
  const float* ln1_b = (const float*)d_in[13];
  const float* ln2_s = (const float*)d_in[14];
  const float* ln2_b = (const float*)d_in[15];
  const float* lnf_s = (const float*)d_in[16];
  const float* lnf_b = (const float*)d_in[17];
  const float* Wlm   = (const float*)d_in[18];
  const float* blm   = (const float*)d_in[19];
  float* logits = (float*)d_out;

  char* ws = (char*)d_ws;
  size_t off = 0;
  auto carve = [&](size_t bytes) -> char* {
    char* p = ws + off;
    off = (off + bytes + 255) & ~(size_t)255;
    return p;
  };
  __bf16* wqkv = (__bf16*)carve((size_t)GPT_L * GPT_D * 3 * GPT_D * 2);
  __bf16* wo   = (__bf16*)carve((size_t)GPT_L * GPT_D * GPT_D * 2);
  __bf16* w1   = (__bf16*)carve((size_t)GPT_L * GPT_D * GPT_F * 2);
  __bf16* w2   = (__bf16*)carve((size_t)GPT_L * GPT_F * GPT_D * 2);
  __bf16* wlm  = (__bf16*)carve((size_t)GPT_D * GPT_V * 2);
  float*  x    = (float*) carve((size_t)ROWS * GPT_D * 4);
  __bf16* hbuf = (__bf16*)carve((size_t)ROWS * GPT_D * 2);
  __bf16* qkv  = (__bf16*)carve((size_t)ROWS * 3 * GPT_D * 2);
  __bf16* aout = (__bf16*)carve((size_t)ROWS * GPT_D * 2);
  __bf16* mlph = (__bf16*)carve((size_t)ROWS * GPT_F * 2);
  (void)ws_size; (void)n_in; (void)in_sizes; (void)out_size;

  auto blocks = [](long long n) { return (unsigned)((n + 255) / 256); };

  // ---- weight prep (bf16) ----
  pack_qkv_w<<<blocks((long long)GPT_L * GPT_D * 3 * GPT_D), 256, 0, stream>>>(
      Wq, Wk, Wv, wqkv);
  cvt_f32_bf16<<<blocks((long long)GPT_L * GPT_D * GPT_D), 256, 0, stream>>>(
      Wo, wo, (long long)GPT_L * GPT_D * GPT_D);
  cvt_f32_bf16<<<blocks((long long)GPT_L * GPT_D * GPT_F), 256, 0, stream>>>(
      W1, w1, (long long)GPT_L * GPT_D * GPT_F);
  cvt_f32_bf16<<<blocks((long long)GPT_L * GPT_F * GPT_D), 256, 0, stream>>>(
      W2, w2, (long long)GPT_L * GPT_F * GPT_D);
  cvt_f32_bf16<<<blocks((long long)GPT_D * GPT_V), 256, 0, stream>>>(
      Wlm, wlm, (long long)GPT_D * GPT_V);

  // ---- embedding ----
  embed_kernel<<<blocks((long long)ROWS * GPT_D), 256, 0, stream>>>(idx, tok, pos, x);

  // ---- transformer layers ----
  for (int l = 0; l < GPT_L; ++l) {
    ln_bf16<<<ROWS, 128, 0, stream>>>(x, ln1_s + l * GPT_D, ln1_b + l * GPT_D, hbuf);
    // qkv = h @ Wqkv  [4096 x 3072] bf16
    gemm_bf16<0, 0, 0, 0><<<dim3(3 * GPT_D / 128, ROWS / 128), 256, 0, stream>>>(
        hbuf, wqkv + (size_t)l * GPT_D * 3 * GPT_D, ROWS, 3 * GPT_D, GPT_D,
        nullptr, nullptr, nullptr, qkv);
    // flash attention
    attn_kernel<<<dim3(GPT_B * GPT_H, GPT_T / 16), 32, 0, stream>>>(qkv, aout);
    // x = x + attn @ Wo + bo   (fp32, residual)
    gemm_bf16<1, 0, 1, 1><<<dim3(GPT_D / 128, ROWS / 128), 256, 0, stream>>>(
        aout, wo + (size_t)l * GPT_D * GPT_D, ROWS, GPT_D, GPT_D,
        bo + l * GPT_D, x, x, nullptr);
    ln_bf16<<<ROWS, 128, 0, stream>>>(x, ln2_s + l * GPT_D, ln2_b + l * GPT_D, hbuf);
    // mlph = relu(h2 @ W1 + b1)  [4096 x 4096] bf16
    gemm_bf16<1, 1, 0, 0><<<dim3(GPT_F / 128, ROWS / 128), 256, 0, stream>>>(
        hbuf, w1 + (size_t)l * GPT_D * GPT_F, ROWS, GPT_F, GPT_D,
        b1 + l * GPT_F, nullptr, nullptr, mlph);
    // x = x + mlph @ W2 + b2   (fp32, residual)
    gemm_bf16<1, 0, 1, 1><<<dim3(GPT_D / 128, ROWS / 128), 256, 0, stream>>>(
        mlph, w2 + (size_t)l * GPT_F * GPT_D, ROWS, GPT_D, GPT_F,
        b2 + l * GPT_D, x, x, nullptr);
  }

  // ---- final LN + LM head ----
  ln_bf16<<<ROWS, 128, 0, stream>>>(x, lnf_s, lnf_b, hbuf);
  gemm_bf16<1, 0, 0, 1><<<dim3(GPT_V / 128, ROWS / 128), 256, 0, stream>>>(
      hbuf, wlm, ROWS, GPT_V, GPT_D, blm, nullptr, logits, nullptr);
}
